// CRF_46170898432426
// MI455X (gfx1250) — matrix operand order repaired and checked by reading, not compiled, not run on hardware
//
#include <hip/hip_runtime.h>
#include <hip/hip_bf16.h>
#include <math.h>
#include <stdint.h>

typedef __attribute__((ext_vector_type(16))) _Float16     v16h;
typedef __attribute__((ext_vector_type(8)))  float        v8f;
typedef __attribute__((ext_vector_type(4)))  unsigned int v4u;
typedef __attribute__((ext_vector_type(4)))  unsigned int u32x4;
typedef __attribute__((ext_vector_type(8)))  int          i32x8;
typedef __attribute__((ext_vector_type(4)))  int          i32x4;

#define B_SZ 512
#define T_SZ 1024
#define N_SZ 64
#define WAVES_PER_BLOCK 4
#define ROWS_PER_WAVE   16

#define LOG2E_F 1.4426950408889634f
#define LN2_F   0.6931471805599453f

#if __has_builtin(__builtin_amdgcn_tensor_load_to_lds) && __has_builtin(__builtin_amdgcn_s_wait_tensorcnt)
#define USE_TDM 1
#else
#define USE_TDM 0
#endif

#if __has_builtin(__builtin_amdgcn_exp2f)
#define FAST_EXP2(x) __builtin_amdgcn_exp2f(x)
#else
#define FAST_EXP2(x) exp2f(x)      // device libm; lowers to native v_exp_f32
#endif
#if __has_builtin(__builtin_amdgcn_logf)
#define FAST_LOG2(x) __builtin_amdgcn_logf(x)   // raw v_log_f32 (log2)
#else
#define FAST_LOG2(x) __log2f(x)
#endif

// One wave = 16 batch rows. alpha kept in registers in WMMA D-tile layout:
//   alphaR[n][i] = alpha[row = hh*8+i][col = 16n + r],  lane = hh*16 + r.
// Recursion step: alpha' = x_t + m + ln2*log2( exp2((alpha-m)*log2e) @ E ),
// E = exp(trans) prepacked once into 8 WMMA B fragments (VGPR-resident).
// x_t tiles stream in via the Tensor Data Mover, double-buffered in LDS.
__global__ __launch_bounds__(WAVES_PER_BLOCK * 32)
void crf_fwd_wmma_kernel(const float* __restrict__ inp,
                         const float* __restrict__ trans,
                         const int*   __restrict__ tags,
                         const int*   __restrict__ lens,
                         float*       __restrict__ out)
{
    __shared__ __align__(16) _Float16 pS[WAVES_PER_BLOCK][ROWS_PER_WAVE][N_SZ];
#if USE_TDM
    __shared__ __align__(16) float    xS[WAVES_PER_BLOCK][2][ROWS_PER_WAVE][N_SZ];
#endif
    __shared__ float lnormS[WAVES_PER_BLOCK][ROWS_PER_WAVE];
    __shared__ int   lenS  [WAVES_PER_BLOCK][ROWS_PER_WAVE];

    const int lane = threadIdx.x & 31;
    const int wave = threadIdx.x >> 5;
    const int b0   = (blockIdx.x * WAVES_PER_BLOCK + wave) * ROWS_PER_WAVE;
    const int r    = lane & 15;
    const int hh   = lane >> 4;

    _Float16 (*pbuf)[N_SZ] = pS[wave];
    float*   lnorm = lnormS[wave];
    int*     lenb  = lenS[wave];

    if (lane < ROWS_PER_WAVE) lenb[lane] = lens[b0 + lane];

    // ---- masked-update thresholds per D-tile row (registers, branch-free) ----
    int tcmp[8];
    #pragma unroll
    for (int i = 0; i < 8; ++i) tcmp[i] = lenb[hh * 8 + i] - 1;

    // ---- alpha = inputs[:,0,:] in D-tile register layout ----
    float alphaR[4][8];
    #pragma unroll
    for (int n = 0; n < 4; ++n)
        #pragma unroll
        for (int i = 0; i < 8; ++i)
            alphaR[n][i] = inp[(size_t)(b0 + hh * 8 + i) * (T_SZ * N_SZ) + 16 * n + r];

    // ---- B fragments once: E = exp(trans), 32x16 f16 WMMA-B layout ----
    v16h Bf[2][4];
    #pragma unroll
    for (int c = 0; c < 2; ++c)
        #pragma unroll
        for (int n = 0; n < 4; ++n) {
            v16h bf;
            const int col = 16 * n + r;
            #pragma unroll
            for (int j = 0; j < 8; ++j) {
                const int k0 = 32 * c + hh * 16 + 2 * j;
                bf[2 * j]     = (_Float16)FAST_EXP2(trans[k0 * N_SZ + col] * LOG2E_F);
                bf[2 * j + 1] = (_Float16)FAST_EXP2(trans[(k0 + 1) * N_SZ + col] * LOG2E_F);
            }
            Bf[c][n] = bf;
        }

#if USE_TDM
    // ---- Tensor Data Mover: 2D descriptor for one 16x64 f32 tile ----
    // tensor: dim0 = T*N elements (stride between batches = T*N), dim1 = B
    i32x8 g1c;
    g1c[0] = (int)(2u << 16);                 // workgroup_mask=0, data_size=4B
    g1c[1] = (int)((65536u & 0xffffu) << 16); // tensor_dim0[15:0] -> bits63:48
    g1c[2] = (int)((65536u >> 16) | (512u << 16)); // dim0[31:16] | tensor_dim1[15:0]
    g1c[3] = (int)(64u << 16);                // tile_dim0 = 64 -> bits127:112
    g1c[4] = 16;                              // tile_dim1 = 16, tile_dim2 = 0
    g1c[5] = 65536;                           // tensor_dim0_stride[31:0] = T*N
    g1c[6] = 0;
    g1c[7] = 0;
    const i32x4 g2c = {0, 0, 0, 0};
    const i32x4 g3c = {0, 0, 0, 0};
    const i32x8 g4c = {0, 0, 0, 0, 0, 0, 0, 0};   // 6-arg toolchain form
    const unsigned xlds0 = (unsigned)(uintptr_t)&xS[wave][0][0][0];
    const unsigned long long gbase =
        (unsigned long long)(uintptr_t)inp + (unsigned long long)b0 * (T_SZ * N_SZ) * 4ull;

    auto tdm_issue = [&](int tt, int bufIdx) {
        const unsigned long long ga = gbase + (unsigned long long)tt * (N_SZ * 4);
        u32x4 g0;
        g0[0] = 1u;                                            // count=1
        g0[1] = xlds0 + (unsigned)bufIdx * (ROWS_PER_WAVE * N_SZ * 4);
        g0[2] = (unsigned)ga;
        g0[3] = (unsigned)((ga >> 32) & 0x01ffffffu) | (2u << 30);  // type=2
        asm volatile("" ::: "memory");   // keep prior LDS reads before overwrite
        __builtin_amdgcn_tensor_load_to_lds(g0, g1c, g2c, g3c, g4c, 0);
    };

    tdm_issue(1, 0);   // prefetch x_1 into buffer 0
#else
    size_t rowOff[8];
    #pragma unroll
    for (int i = 0; i < 8; ++i)
        rowOff[i] = (size_t)(b0 + hh * 8 + i) * (size_t)(T_SZ * N_SZ) + (size_t)r;
#endif

    // =================== forward recursion over T ===================
    for (int t = 0; t < T_SZ - 1; ++t) {
#if USE_TDM
        const int cur = t & 1;
        if (t < T_SZ - 2) {
            tdm_issue(t + 2, cur ^ 1);               // double-buffered prefetch
            __builtin_amdgcn_s_wait_tensorcnt(1);    // oldest (x_{t+1}) complete
        } else {
            __builtin_amdgcn_s_wait_tensorcnt(0);
        }
        asm volatile("" ::: "memory");               // order LDS reads after wait
        // plain LDS pointer -> ds_load_b32 with immediate offsets
        const float* xcur = &xS[wave][cur][0][0] + (hh * 8) * N_SZ + r;
#endif

        // ---- update gate per row: 1.0 while recursion active, else 0.0 ----
        // kept opaque so the compiler cannot re-form a select and sink the
        // log/load into per-element exec-masked branches.
        float mk[8];
        #pragma unroll
        for (int i = 0; i < 8; ++i) {
            float g = (t < tcmp[i]) ? 1.0f : 0.0f;   // v_cndmask of constants
            asm("" : "+v"(g));
            mk[i] = g;
        }

        // ---- per-row max via 16-lane butterfly (rows stay in lane groups) ----
        float m[8];
        #pragma unroll
        for (int i = 0; i < 8; ++i) {
            float v = fmaxf(fmaxf(alphaR[0][i], alphaR[1][i]),
                            fmaxf(alphaR[2][i], alphaR[3][i]));
            v = fmaxf(v, __shfl_xor(v, 1));
            v = fmaxf(v, __shfl_xor(v, 2));
            v = fmaxf(v, __shfl_xor(v, 4));
            v = fmaxf(v, __shfl_xor(v, 8));
            m[i] = v;
        }

        // ---- p = exp(alpha - m) -> f16, scattered to row-major LDS ----
        _Float16* pw = &pbuf[0][0] + hh * 512 + r;   // row = hh*8+i, col = 16n+r
        #pragma unroll
        for (int i = 0; i < 8; ++i)
            #pragma unroll
            for (int n = 0; n < 4; ++n)
                pw[i * 64 + n * 16] =
                    (_Float16)FAST_EXP2((alphaR[n][i] - m[i]) * LOG2E_F);

        // ---- A fragments (16x32 f16 layout) via ds_load_b128 ----
        const char* pb = (const char*)&pbuf[0][0];
        union { v16h h; v4u u[2]; } a0, a1;
        const int base = r * 128 + hh * 32;
        a0.u[0] = *(const v4u*)(pb + base);
        a0.u[1] = *(const v4u*)(pb + base + 16);
        a1.u[0] = *(const v4u*)(pb + base + 64);
        a1.u[1] = *(const v4u*)(pb + base + 80);

        // ---- 4 tiles x 2 K-chunks of WMMA, arithmetic-masked update ----
        #pragma unroll
        for (int n = 0; n < 4; ++n) {
            v8f acc = {};
            acc = __builtin_amdgcn_wmma_f32_16x16x32_f16(
                false, a0.h, false, Bf[0][n], (short)0, acc, false, false);
            acc = __builtin_amdgcn_wmma_f32_16x16x32_f16(
                false, a1.h, false, Bf[1][n], (short)0, acc, false, false);
            #pragma unroll
            for (int i = 0; i < 8; ++i) {
#if USE_TDM
                const float x = xcur[i * N_SZ + 16 * n];
#else
                const float x = inp[rowOff[i] + (size_t)(t + 1) * N_SZ + 16 * n];
#endif
                const float nv = x + m[i] + LN2_F * FAST_LOG2(acc[i]);
                alphaR[n][i] += mk[i] * (nv - alphaR[n][i]);   // v_sub + v_fmac
            }
        }
    }

    // =================== log_norm = logsumexp(alpha, axis=tags) ===================
    #pragma unroll
    for (int i = 0; i < 8; ++i) {
        float v = fmaxf(fmaxf(alphaR[0][i], alphaR[1][i]),
                        fmaxf(alphaR[2][i], alphaR[3][i]));
        v = fmaxf(v, __shfl_xor(v, 1));
        v = fmaxf(v, __shfl_xor(v, 2));
        v = fmaxf(v, __shfl_xor(v, 4));
        v = fmaxf(v, __shfl_xor(v, 8));
        float s = FAST_EXP2((alphaR[0][i] - v) * LOG2E_F)
                + FAST_EXP2((alphaR[1][i] - v) * LOG2E_F)
                + FAST_EXP2((alphaR[2][i] - v) * LOG2E_F)
                + FAST_EXP2((alphaR[3][i] - v) * LOG2E_F);
        s += __shfl_xor(s, 1);
        s += __shfl_xor(s, 2);
        s += __shfl_xor(s, 4);
        s += __shfl_xor(s, 8);
        if (r == i) lnorm[hh * 8 + i] = v + LN2_F * FAST_LOG2(s);
    }

    // =================== sequence scores (unary + binary gathers) ===================
    const int b   = b0 + r;
    const int len = lenb[r];
    float s = 0.f;
    const int tlo = hh * (T_SZ / 2), thi = tlo + (T_SZ / 2);
    for (int t = tlo; t < thi; ++t) {
        if (t < len) {
            const int tg = tags[(size_t)b * T_SZ + t];
            s += inp[(size_t)b * (T_SZ * N_SZ) + (size_t)t * N_SZ + tg];
            if (t >= 1) {
                const int tp = tags[(size_t)b * T_SZ + t - 1];
                s += trans[tp * N_SZ + tg];
            }
        }
    }
    s += __shfl_xor(s, 16);

    if (hh == 0) out[b] = s - lnorm[r];
}

extern "C" void kernel_launch(void* const* d_in, const int* in_sizes, int n_in,
                              void* d_out, int out_size, void* d_ws, size_t ws_size,
                              hipStream_t stream)
{
    const float* inp   = (const float*)d_in[0];   // (B, T, N) f32
    const float* trans = (const float*)d_in[1];   // (N, N)   f32
    const int*   tags  = (const int*)d_in[2];     // (B, T)   i32
    const int*   lens  = (const int*)d_in[3];     // (B,)     i32
    float*       out   = (float*)d_out;           // (B,)     f32

    const int blocks = B_SZ / (WAVES_PER_BLOCK * ROWS_PER_WAVE);   // 8
    crf_fwd_wmma_kernel<<<blocks, WAVES_PER_BLOCK * 32, 0, stream>>>(
        inp, trans, tags, lens, out);
}
